// LSTM_for_NSP_87007447482507
// MI455X (gfx1250) — matrix operand order repaired and checked by reading, not compiled, not run on hardware
//
#include <hip/hip_runtime.h>
#include <cstdint>

// ---------------------------------------------------------------------------
// match-LSTM (NLI) on gfx1250.  All matmuls via v_wmma_f32_16x16x32_f16.
// Weights packed once to f16 WMMA-B fragments (L2-resident, ~14MB << 192MB L2);
// recurrent h written straight into WMMA-A fragment layout by each epilogue.
// Recurrences run inside two persistent kernels with device-scope atomic
// barriers (~15 launches total instead of ~230).
// NOTE: all k-tile loops are '#pragma unroll 1' -- with constant bounds the
// compiler otherwise fully unrolls ~104 WMMAs/step, spills the B fragments to
// scratch, and every WMMA reloads from scratch (seen in round-2 disasm).
// Rolled loops keep ~80 live VGPRs and stream B straight from L2.
// ---------------------------------------------------------------------------

typedef _Float16 half_t;
typedef __attribute__((ext_vector_type(16))) _Float16 v16h;
typedef __attribute__((ext_vector_type(8)))  float    v8f;

#define B_   64
#define H_   512
#define PL_  64
#define TL_  32
#define E_   300
#define EKT  10     // ceil(300/32) k-tiles for embeddings (zero padded)
#define HKT  16     // 512/32 k-tiles for H
#define C_   2
#define FRH  512    // halves per fragment
#define HFR  (4 * HKT * FRH)   // one [64 rows][512 cols] A-fragment set

// ----------------------------- device helpers ------------------------------

__device__ __forceinline__ v16h ldfrag(const half_t* p) {
  return *reinterpret_cast<const v16h*>(p);
}
__device__ __forceinline__ float sigmoidf_(float x) { return 1.f / (1.f + __expf(-x)); }

// Offset (halves) of element (row,k) in an A-frag buffer [mt][kt][512].
// 16-bit A 16x32 layout (contiguous): lanes 0-15 hold K {0..15}, lanes 16-31
// hold K {16..31}; 16 halves per lane.
__device__ __forceinline__ long long afrag_off(int row, int k, int KT) {
  int mt = row >> 4, kt = k >> 5;
  int kk = k & 31;
  int l  = (row & 15) + (((kk >> 4) & 1) << 4);
  int e  = kk & 15;
  return ((long long)(mt * KT + kt) << 9) + (l << 4) + e;
}

// Device-wide barrier over co-resident blocks (monotonic counter).
__device__ __forceinline__ void grid_barrier(unsigned* cnt, unsigned target) {
  __syncthreads();
  __threadfence();                                   // release our stores (DEV)
  if (threadIdx.x == 0) {
    __hip_atomic_fetch_add(cnt, 1u, __ATOMIC_RELEASE, __HIP_MEMORY_SCOPE_AGENT);
    while (__hip_atomic_load(cnt, __ATOMIC_ACQUIRE, __HIP_MEMORY_SCOPE_AGENT) < target)
      __builtin_amdgcn_s_sleep(2);
  }
  __syncthreads();
  __threadfence();                                   // acquire: invalidate near caches
}

// ---------------- shared WMMA tile worker: one LSTM step tile --------------
// Wave (mt,jt) computes the 16x16 tiles of all four gates for batch tile mt
// and H-column tile jt, then the c/h elementwise update.
__device__ __forceinline__ void lstm_tile(
    int l, int mt, int jt,
    const half_t* __restrict__ x1, int kt1,
    const half_t* __restrict__ x2, int kt2,
    const half_t* __restrict__ hin,
    const half_t* __restrict__ wih,      // [128 nt][kt1+kt2][512]
    const half_t* __restrict__ whh,      // [128 nt][16][512]
    const float* __restrict__ bih, const float* __restrict__ bhh,
    float* __restrict__ c_state, half_t* __restrict__ hout,
    float* __restrict__ hs_out, half_t* __restrict__ hsfrag_out,
    const int* __restrict__ seq_len, int t,
    const int* __restrict__ hyp_len, float* __restrict__ h_last, int kstep) {
  int KTx = kt1 + kt2;
  v8f acc[4] = {};
#pragma unroll 1
  for (int kt = 0; kt < kt1; ++kt) {
    v16h a = ldfrag(x1 + (((long long)(mt * kt1 + kt)) << 9) + (l << 4));
#pragma unroll
    for (int g = 0; g < 4; ++g) {
      int nt = g * 32 + jt;
      v16h b = ldfrag(wih + (((long long)(nt * KTx + kt)) << 9) + (l << 4));
      acc[g] = __builtin_amdgcn_wmma_f32_16x16x32_f16(false, a, false, b,
                                                      (short)0, acc[g], false, false);
    }
  }
#pragma unroll 1
  for (int kt = 0; kt < kt2; ++kt) {
    v16h a = ldfrag(x2 + (((long long)(mt * kt2 + kt)) << 9) + (l << 4));
#pragma unroll
    for (int g = 0; g < 4; ++g) {
      int nt = g * 32 + jt;
      v16h b = ldfrag(wih + (((long long)(nt * KTx + kt1 + kt)) << 9) + (l << 4));
      acc[g] = __builtin_amdgcn_wmma_f32_16x16x32_f16(false, a, false, b,
                                                      (short)0, acc[g], false, false);
    }
  }
#pragma unroll 1
  for (int kt = 0; kt < HKT; ++kt) {
    v16h a = ldfrag(hin + (((long long)(mt * HKT + kt)) << 9) + (l << 4));
#pragma unroll
    for (int g = 0; g < 4; ++g) {
      int nt = g * 32 + jt;
      v16h b = ldfrag(whh + (((long long)(nt * HKT + kt)) << 9) + (l << 4));
      acc[g] = __builtin_amdgcn_wmma_f32_16x16x32_f16(false, a, false, b,
                                                      (short)0, acc[g], false, false);
    }
  }
  int n = l & 15;
  int mbase = (l >= 16) ? 8 : 0;
  int j = jt * 16 + n;
  float b_i = bih[j]        + bhh[j];
  float b_f = bih[512 + j]  + bhh[512 + j];
  float b_g = bih[1024 + j] + bhh[1024 + j];
  float b_o = bih[1536 + j] + bhh[1536 + j];
#pragma unroll
  for (int r = 0; r < 8; ++r) {
    int bb = mt * 16 + mbase + r;
    float iv = acc[0][r] + b_i;
    float fv = acc[1][r] + b_f;
    float gv = acc[2][r] + b_g;
    float ov = acc[3][r] + b_o;
    float c_old = c_state[bb * H_ + j];
    float c_new = sigmoidf_(fv) * c_old + sigmoidf_(iv) * tanhf(gv);
    float h = sigmoidf_(ov) * tanhf(c_new);
    c_state[bb * H_ + j] = c_new;
    hout[afrag_off(bb, j, HKT)] = (half_t)h;
    if (hs_out || hsfrag_out) {
      float hm = (t < seq_len[bb]) ? h : 0.f;
      if (hs_out)     hs_out[bb * H_ + j] = hm;
      if (hsfrag_out) hsfrag_out[afrag_off(bb, j, HKT)] = (half_t)hm;
    }
    if (h_last) {
      if (kstep + 1 == hyp_len[bb]) h_last[bb * H_ + j] = h;
    }
  }
}

// ----------------------- weight packing (B fragments) ----------------------
__global__ void pack_b_kernel(const float* __restrict__ W, half_t* __restrict__ out,
                              int N, int Kreal, int KT) {
  long long idx = (long long)blockIdx.x * blockDim.x + threadIdx.x;
  long long total = (long long)(N >> 4) * KT * 512;
  if (idx >= total) return;
  int e = (int)(idx & 15);
  int l = (int)((idx >> 4) & 31);
  long long frag = idx >> 9;
  int kt = (int)(frag % KT);
  int nt = (int)(frag / KT);
  int n = nt * 16 + (l & 15);
  int k = kt * 32 + e + ((l >= 16) ? 16 : 0);
  float v = (k < Kreal) ? W[(long long)n * Kreal + k] : 0.f;
  out[idx] = (half_t)v;
}

// ------------------- embedding gather into A fragments ---------------------
__global__ void embed_pack_kernel(const int* __restrict__ tok,
                                  const float* __restrict__ embed,
                                  half_t* __restrict__ out, int T) {
  long long idx = (long long)blockIdx.x * blockDim.x + threadIdx.x;
  long long total = (long long)T * 4 * EKT * 512;
  if (idx >= total) return;
  int e = (int)(idx & 15);
  int l = (int)((idx >> 4) & 31);
  long long frag = idx >> 9;
  int kt = (int)(frag % EKT);
  long long r = frag / EKT;
  int mt = (int)(r & 3);
  int t  = (int)(r >> 2);
  int b = mt * 16 + (l & 15);
  int k = kt * 32 + e + ((l >= 16) ? 16 : 0);
  float v = 0.f;
  if (k < E_) {
    int id = tok[t * B_ + b];
    v = embed[(long long)id * E_ + k];
  }
  out[idx] = (half_t)v;
}

// --------------- persistent encoder: premise + hypothesis LSTMs ------------
// 32 blocks: blocks 0-15 premise (64 steps), 16-31 hypothesis (32 steps).
__global__ void encoder_kernel(
    const half_t* __restrict__ xf_p, const half_t* __restrict__ wf_pih,
    const half_t* __restrict__ wf_phh,
    const float* __restrict__ pbih, const float* __restrict__ pbhh,
    float* __restrict__ c_p, half_t* __restrict__ hp0, half_t* __restrict__ hp1,
    float* __restrict__ hs_f32, half_t* __restrict__ hsfrag,
    const int* __restrict__ prem_len,
    const half_t* __restrict__ xf_h, const half_t* __restrict__ wf_hih,
    const half_t* __restrict__ wf_hhh,
    const float* __restrict__ hbih, const float* __restrict__ hbhh,
    float* __restrict__ c_h, half_t* __restrict__ hh0, half_t* __restrict__ hh1,
    half_t* __restrict__ htfrag, const int* __restrict__ hyp_len,
    unsigned* __restrict__ cnt) {
  int l  = threadIdx.x & 31;
  int grp = blockIdx.x >> 4;           // 0: premise, 1: hypothesis
  int lb  = blockIdx.x & 15;
  int w   = lb * 8 + (threadIdx.x >> 5);
  int mt = w >> 5, jt = w & 31;

  int T                 = grp ? TL_     : PL_;
  const half_t* xbase   = grp ? xf_h    : xf_p;
  const half_t* wih     = grp ? wf_hih  : wf_pih;
  const half_t* whh     = grp ? wf_hhh  : wf_phh;
  const float*  bih     = grp ? hbih    : pbih;
  const float*  bhh     = grp ? hbhh    : pbhh;
  float*        c_state = grp ? c_h     : c_p;
  half_t*       hA      = grp ? hh0     : hp0;
  half_t*       hB      = grp ? hh1     : hp1;
  const int*    slen    = grp ? hyp_len : prem_len;
  unsigned*     bar     = cnt + grp;

#pragma unroll 1
  for (int t = 0; t < T; ++t) {
    const half_t* x1 = xbase + (long long)t * 4 * EKT * FRH;
    const half_t* hin  = (t & 1) ? hB : hA;
    half_t*       hout = (t & 1) ? hA : hB;
    float*  hsf = grp ? nullptr : (hs_f32 + (long long)t * B_ * H_);
    half_t* hff = grp ? (htfrag + (long long)t * HFR)
                      : (hsfrag + (long long)t * HFR);
    // prefetch next step's x fragments for this mt slice (known one step early)
    if (t + 1 < T) {
      const char* nx = (const char*)(xbase + ((long long)(t + 1) * 4 + mt) * EKT * FRH);
      __builtin_prefetch(nx + (long long)l * 640, 0, 0);   // cover 10KB slice
    }
    lstm_tile(l, mt, jt, x1, EKT, nullptr, 0, hin, wih, whh, bih, bhh,
              c_state, hout, hsf, hff, slen, t, nullptr, nullptr, 0);
    grid_barrier(bar, 16u * (unsigned)(t + 1));
  }
}

// ----------------------- hs_Ws = h_s @ Ws.T (wide GEMM) --------------------
__global__ void gemm_frag_kernel(const half_t* __restrict__ a1, int KT1,
                                 const half_t* __restrict__ b1,
                                 float* __restrict__ out, int MT, int NT) {
  int l = threadIdx.x & 31;
  int w = blockIdx.x * (blockDim.x >> 5) + (threadIdx.x >> 5);
  if (w >= MT * NT) return;
  int mt = w / NT, nt = w % NT;
  v8f acc = {};
#pragma unroll 1
  for (int kt = 0; kt < KT1; ++kt) {
    v16h a = ldfrag(a1 + (((long long)(mt * KT1 + kt)) << 9) + (l << 4));
    v16h b = ldfrag(b1 + (((long long)(nt * KT1 + kt)) << 9) + (l << 4));
    acc = __builtin_amdgcn_wmma_f32_16x16x32_f16(false, a, false, b,
                                                 (short)0, acc, false, false);
  }
  int ncol = nt * 16 + (l & 15);
  int mbase = mt * 16 + ((l >= 16) ? 8 : 0);
  long long ldc = (long long)NT * 16;
#pragma unroll
  for (int r = 0; r < 8; ++r)
    out[(long long)(mbase + r) * ldc + ncol] = acc[r];
}

// -------------- persistent match-LSTM: 32 iters x 4 phases + FC ------------
// 16 blocks x 8 waves = 128 waves.
__global__ void match_kernel(
    const half_t* __restrict__ htfrag, half_t* __restrict__ hm0,
    half_t* __restrict__ hm1, half_t* __restrict__ afrag,
    const half_t* __restrict__ wf_Wt, const half_t* __restrict__ wf_Wm,
    const half_t* __restrict__ wf_mih, const half_t* __restrict__ wf_mhh,
    const float* __restrict__ mbih, const float* __restrict__ mbhh,
    float* __restrict__ c_match, float* __restrict__ h_last,
    const float* __restrict__ hsWs, float* __restrict__ lin,
    float* __restrict__ e_buf, const float* __restrict__ hs_f32,
    const float* __restrict__ w_e, const int* __restrict__ hyp_len,
    const float* __restrict__ fc_W, const float* __restrict__ fc_b,
    float* __restrict__ out, unsigned* __restrict__ cnt) {
  int l = threadIdx.x & 31;
  int w = blockIdx.x * 8 + (threadIdx.x >> 5);
  unsigned ep = 0;

#pragma unroll 1
  for (int k = 0; k < TL_; ++k) {
    const half_t* htk  = htfrag + (long long)k * HFR;
    const half_t* hmin = (k & 1) ? hm1 : hm0;
    half_t*       hmout = (k & 1) ? hm0 : hm1;

    // -- phase 1: lin = h_t_k @ Wt.T + h_m @ Wm.T  (WMMA, 1 tile/wave) -----
    {
      int mt = w >> 5, nt = w & 31;
      v8f acc = {};
#pragma unroll 1
      for (int kt = 0; kt < HKT; ++kt) {
        v16h a = ldfrag(htk + (((long long)(mt * HKT + kt)) << 9) + (l << 4));
        v16h b = ldfrag(wf_Wt + (((long long)(nt * HKT + kt)) << 9) + (l << 4));
        acc = __builtin_amdgcn_wmma_f32_16x16x32_f16(false, a, false, b,
                                                     (short)0, acc, false, false);
      }
#pragma unroll 1
      for (int kt = 0; kt < HKT; ++kt) {
        v16h a = ldfrag(hmin + (((long long)(mt * HKT + kt)) << 9) + (l << 4));
        v16h b = ldfrag(wf_Wm + (((long long)(nt * HKT + kt)) << 9) + (l << 4));
        acc = __builtin_amdgcn_wmma_f32_16x16x32_f16(false, a, false, b,
                                                     (short)0, acc, false, false);
      }
      int ncol = nt * 16 + (l & 15);
      int mrow = mt * 16 + ((l >= 16) ? 8 : 0);
#pragma unroll
      for (int r = 0; r < 8; ++r) lin[(mrow + r) * H_ + ncol] = acc[r];
    }
    grid_barrier(cnt, 16u * ++ep);

    // -- phase 2: e[p,b] = sum_h tanh(hsWs + lin) * w_e  (32 pairs/wave) ---
#pragma unroll 1
    for (int i = 0; i < 32; ++i) {
      int idx = w * 32 + i;
      int p = idx >> 6, b = idx & 63;
      const float* hw  = hsWs + ((long long)(p * B_ + b)) * H_;
      const float* lb2 = lin + (long long)b * H_;
      float s = 0.f;
#pragma unroll 1
      for (int h = l; h < H_; h += 32) s += tanhf(hw[h] + lb2[h]) * w_e[h];
#pragma unroll
      for (int off = 16; off > 0; off >>= 1) s += __shfl_xor(s, off, 32);
      if (l == 0) e_buf[p * B_ + b] = s;
    }
    grid_barrier(cnt, 16u * ++ep);

    // -- phase 3: softmax over p + context a -> match-input A frags --------
    if (w < B_) {
      int b = w;
      float a0 = e_buf[l * B_ + b];           // p = l
      float a1 = e_buf[(l + 32) * B_ + b];    // p = l + 32
      float m = fmaxf(a0, a1);
#pragma unroll
      for (int off = 16; off > 0; off >>= 1) m = fmaxf(m, __shfl_xor(m, off, 32));
      float e0 = __expf(a0 - m), e1 = __expf(a1 - m);
      float s = e0 + e1;
#pragma unroll
      for (int off = 16; off > 0; off >>= 1) s += __shfl_xor(s, off, 32);
      float inv = 1.f / s;
      float acc[16];
#pragma unroll
      for (int i = 0; i < 16; ++i) acc[i] = 0.f;
#pragma unroll 1
      for (int p = 0; p < PL_; ++p) {
        float av = __shfl((p < 32) ? e0 : e1, p & 31, 32);
        const float* row = hs_f32 + ((long long)(p * B_ + b)) * H_ + l * 16;
#pragma unroll
        for (int i = 0; i < 16; ++i) acc[i] += av * row[i];
      }
#pragma unroll
      for (int i = 0; i < 16; ++i)
        afrag[afrag_off(b, l * 16 + i, HKT)] = (half_t)(acc[i] * inv);
    }
    grid_barrier(cnt, 16u * ++ep);

    // -- phase 4: match LSTM cell, input [a | h_t_k] (K=1024) --------------
    lstm_tile(l, w >> 5, w & 31, afrag, HKT, htk, HKT, hmin,
              wf_mih, wf_mhh, mbih, mbhh, c_match, hmout,
              nullptr, nullptr, nullptr, k, hyp_len, h_last, k);
    grid_barrier(cnt, 16u * ++ep);
  }

  // -- final classifier (block 0) ------------------------------------------
  if (blockIdx.x == 0 && threadIdx.x < B_ * C_) {
    int b = threadIdx.x >> 1, c = threadIdx.x & 1;
    float s = fc_b[c];
#pragma unroll 1
    for (int h = 0; h < H_; ++h) s += h_last[b * H_ + h] * fc_W[c * H_ + h];
    out[b * C_ + c] = s;
  }
}

// ------------------------------- launcher ----------------------------------

extern "C" void kernel_launch(void* const* d_in, const int* in_sizes, int n_in,
                              void* d_out, int out_size, void* d_ws, size_t ws_size,
                              hipStream_t stream) {
  (void)in_sizes; (void)n_in; (void)out_size; (void)ws_size;
  const int*   premise        = (const int*)  d_in[0];
  const int*   premise_len    = (const int*)  d_in[1];
  const int*   hypothesis     = (const int*)  d_in[2];
  const int*   hypothesis_len = (const int*)  d_in[3];
  const float* embed          = (const float*)d_in[4];
  const float* prem_Wih  = (const float*)d_in[5];
  const float* prem_Whh  = (const float*)d_in[6];
  const float* prem_bih  = (const float*)d_in[7];
  const float* prem_bhh  = (const float*)d_in[8];
  const float* hypo_Wih  = (const float*)d_in[9];
  const float* hypo_Whh  = (const float*)d_in[10];
  const float* hypo_bih  = (const float*)d_in[11];
  const float* hypo_bhh  = (const float*)d_in[12];
  const float* match_Wih = (const float*)d_in[13];
  const float* match_Whh = (const float*)d_in[14];
  const float* match_bih = (const float*)d_in[15];
  const float* match_bhh = (const float*)d_in[16];
  const float* Ws   = (const float*)d_in[17];
  const float* Wt   = (const float*)d_in[18];
  const float* Wm   = (const float*)d_in[19];
  const float* w_e  = (const float*)d_in[20];
  const float* fc_W = (const float*)d_in[21];
  const float* fc_b = (const float*)d_in[22];
  float* out = (float*)d_out;

  // ---- workspace carve-up (256B aligned) ----
  char* ws = (char*)d_ws;
  size_t off = 0;
  auto alloc_h = [&](size_t n) { half_t* p = (half_t*)(ws + off);
                                 off += ((n * 2 + 255) & ~(size_t)255); return p; };
  auto alloc_f = [&](size_t n) { float* p = (float*)(ws + off);
                                 off += ((n * 4 + 255) & ~(size_t)255); return p; };

  half_t* wf_pih = alloc_h(128 * EKT * FRH);
  half_t* wf_phh = alloc_h(128 * HKT * FRH);
  half_t* wf_hih = alloc_h(128 * EKT * FRH);
  half_t* wf_hhh = alloc_h(128 * HKT * FRH);
  half_t* wf_mih = alloc_h(128 * 32 * FRH);
  half_t* wf_mhh = alloc_h(128 * HKT * FRH);
  half_t* wf_Ws  = alloc_h(32 * HKT * FRH);
  half_t* wf_Wt  = alloc_h(32 * HKT * FRH);
  half_t* wf_Wm  = alloc_h(32 * HKT * FRH);
  half_t* xf_p   = alloc_h((size_t)PL_ * 4 * EKT * FRH);
  half_t* xf_h   = alloc_h((size_t)TL_ * 4 * EKT * FRH);
  half_t* hp0 = alloc_h(HFR); half_t* hp1 = alloc_h(HFR);   // premise h ping-pong
  half_t* hh0 = alloc_h(HFR); half_t* hh1 = alloc_h(HFR);   // hypothesis h ping-pong
  half_t* htfrag = alloc_h((size_t)TL_ * HFR);              // masked h_t per step
  half_t* hsfrag = alloc_h((size_t)PL_ * HFR);              // masked h_s (4096 rows)
  half_t* hm0 = alloc_h(HFR); half_t* hm1 = alloc_h(HFR);   // match h ping-pong
  half_t* afrag  = alloc_h(HFR);                            // context a
  float* c_p     = alloc_f((size_t)B_ * H_);
  float* c_h     = alloc_f((size_t)B_ * H_);
  float* c_match = alloc_f((size_t)B_ * H_);
  float* h_lastb = alloc_f((size_t)B_ * H_);
  float* hs_f32  = alloc_f((size_t)PL_ * B_ * H_);
  float* hsWs    = alloc_f((size_t)PL_ * B_ * H_);
  float* lin     = alloc_f((size_t)B_ * H_);
  float* e_buf   = alloc_f((size_t)PL_ * B_);
  unsigned* cnts = (unsigned*)alloc_f(64);                  // barrier counters

  auto blocks_for = [](long long total) { return (unsigned)((total + 255) / 256); };

  // ---- 1. pack weights to f16 B fragments ----
  pack_b_kernel<<<blocks_for(128LL * EKT * FRH), 256, 0, stream>>>(prem_Wih,  wf_pih, 2048, E_,   EKT);
  pack_b_kernel<<<blocks_for(128LL * HKT * FRH), 256, 0, stream>>>(prem_Whh,  wf_phh, 2048, H_,   HKT);
  pack_b_kernel<<<blocks_for(128LL * EKT * FRH), 256, 0, stream>>>(hypo_Wih,  wf_hih, 2048, E_,   EKT);
  pack_b_kernel<<<blocks_for(128LL * HKT * FRH), 256, 0, stream>>>(hypo_Whh,  wf_hhh, 2048, H_,   HKT);
  pack_b_kernel<<<blocks_for(128LL * 32  * FRH), 256, 0, stream>>>(match_Wih, wf_mih, 2048, 1024, 32);
  pack_b_kernel<<<blocks_for(128LL * HKT * FRH), 256, 0, stream>>>(match_Whh, wf_mhh, 2048, H_,   HKT);
  pack_b_kernel<<<blocks_for(32LL  * HKT * FRH), 256, 0, stream>>>(Ws, wf_Ws, H_, H_, HKT);
  pack_b_kernel<<<blocks_for(32LL  * HKT * FRH), 256, 0, stream>>>(Wt, wf_Wt, H_, H_, HKT);
  pack_b_kernel<<<blocks_for(32LL  * HKT * FRH), 256, 0, stream>>>(Wm, wf_Wm, H_, H_, HKT);

  // ---- 2. embedding gather into A fragments ----
  embed_pack_kernel<<<blocks_for((long long)PL_ * 4 * EKT * FRH), 256, 0, stream>>>(premise,    embed, xf_p, PL_);
  embed_pack_kernel<<<blocks_for((long long)TL_ * 4 * EKT * FRH), 256, 0, stream>>>(hypothesis, embed, xf_h, TL_);

  // ---- 3. zero initial state + barrier counters ----
  hipMemsetAsync(hp0, 0, HFR * sizeof(half_t), stream);
  hipMemsetAsync(hh0, 0, HFR * sizeof(half_t), stream);
  hipMemsetAsync(hm0, 0, HFR * sizeof(half_t), stream);
  hipMemsetAsync(c_p,     0, (size_t)B_ * H_ * sizeof(float), stream);
  hipMemsetAsync(c_h,     0, (size_t)B_ * H_ * sizeof(float), stream);
  hipMemsetAsync(c_match, 0, (size_t)B_ * H_ * sizeof(float), stream);
  hipMemsetAsync(h_lastb, 0, (size_t)B_ * H_ * sizeof(float), stream);
  hipMemsetAsync(cnts,    0, 256, stream);

  // ---- 4. persistent encoders: premise (blocks 0-15) + hypothesis (16-31) --
  encoder_kernel<<<32, 256, 0, stream>>>(
      xf_p, wf_pih, wf_phh, prem_bih, prem_bhh, c_p, hp0, hp1,
      hs_f32, hsfrag, premise_len,
      xf_h, wf_hih, wf_hhh, hypo_bih, hypo_bhh, c_h, hh0, hh1,
      htfrag, hypothesis_len, cnts);

  // ---- 5. hs_Ws = h_s @ Ws.T  ([4096,512] x [512,512], 8192 waves) ----
  gemm_frag_kernel<<<blocks_for((long long)256 * 32 * 32), 256, 0, stream>>>(
      hsfrag, HKT, wf_Ws, hsWs, 256, 32);

  // ---- 6. persistent match-LSTM (32 iters x 4 phases) + classifier ----
  match_kernel<<<16, 256, 0, stream>>>(
      htfrag, hm0, hm1, afrag, wf_Wt, wf_Wm, wf_mih, wf_mhh,
      match_bih, match_bhh, c_match, h_lastb, hsWs, lin, e_buf,
      hs_f32, w_e, hypothesis_len, fc_W, fc_b, out, cnts + 2);
}
